// SliceAttention_32452772888732
// MI455X (gfx1250) — matrix operand order repaired and checked by reading, not compiled, not run on hardware
//
#include <hip/hip_runtime.h>
#include <hip/hip_bf16.h>

// ---- problem constants (reference: B=2, S=8192, E=128) ----
#define BATCH 2
#define SEQ   8192
#define EMB   128
#define QKV_BYTES (BATCH * SEQ * EMB * 2)   // bf16 plane: 4 MiB

typedef __attribute__((ext_vector_type(16))) __bf16 v16bf;
typedef __attribute__((ext_vector_type(8)))  __bf16 v8bf;
typedef __attribute__((ext_vector_type(8)))  float  v8f;

// ---------------------------------------------------------------------------
// Fragment loaders matching CDNA5 WMMA VGPR layouts (CDNA5 ISA):
//  16-bit A 16x32: lanes 0-15 M=lane, K={kb+0..7, kb+16..23};
//                  lanes 16-31 M=lane-16, K={kb+8..15, kb+24..31}
//  16-bit B 32x16: lanes 0-15 N=lane, K={kb+0..7, kb+16..23};
//                  lanes 16-31 N=lane-16, K={kb+8..15, kb+24..31}
// ---------------------------------------------------------------------------

__device__ __forceinline__ v16bf a_frag_f32(const float* __restrict__ base,
                                            int ld, int kb, int lane) {
  const int m = lane & 15;
  const int h = (lane >> 4) & 1;
  const float* p = base + m * ld + kb + h * 8;
  v16bf r;
#pragma unroll
  for (int j = 0; j < 8; ++j) r[j] = (__bf16)p[j];
#pragma unroll
  for (int j = 0; j < 8; ++j) r[8 + j] = (__bf16)p[16 + j];
  return r;
}

__device__ __forceinline__ v16bf b_frag_f32(const float* __restrict__ base,
                                            int ld, int n0, int kb, int lane) {
  const int n  = lane & 15;
  const int kh = (lane >> 4) & 1;
  const float* p = base + (n0 + n) * ld + kb + kh * 8;
  v16bf r;
#pragma unroll
  for (int j = 0; j < 16; ++j) r[j] = (__bf16)p[(j & 7) + (j >> 3) * 16];
  return r;
}

__device__ __forceinline__ v16bf a_frag_bf16(const __bf16* base,
                                             int ld, int kb, int lane) {
  const int m = lane & 15;
  const int h = (lane >> 4) & 1;
  const __bf16* p = base + m * ld + kb + h * 8;
  v8bf lo = *(const v8bf*)p;
  v8bf hi = *(const v8bf*)(p + 16);
  v16bf r;
#pragma unroll
  for (int j = 0; j < 8; ++j) { r[j] = lo[j]; r[8 + j] = hi[j]; }
  return r;
}

__device__ __forceinline__ v16bf b_frag_bf16(const __bf16* base,
                                             int ld, int n0, int kb, int lane) {
  const int n  = lane & 15;
  const int kh = (lane >> 4) & 1;
  const __bf16* p = base + (n0 + n) * ld + kb + kh * 8;
  v8bf lo = *(const v8bf*)p;
  v8bf hi = *(const v8bf*)(p + 16);
  v16bf r;
#pragma unroll
  for (int j = 0; j < 8; ++j) { r[j] = lo[j]; r[8 + j] = hi[j]; }
  return r;
}

__device__ __forceinline__ v8f wmma_bf16(v16bf a, v16bf b, v8f c) {
  return __builtin_amdgcn_wmma_f32_16x16x32_bf16(
      /*neg_a=*/false, a, /*neg_b=*/false, b,
      /*c_mod=*/(short)0, c, /*reuse_a=*/false, /*reuse_b=*/false);
}

// Async Global -> LDS copy, 16 bytes (tracked by ASYNCcnt).
// Inline asm per CDNA5 ISA (portable across toolchains).
__device__ __forceinline__ void async_copy_b128(unsigned int lds_addr,
                                                const void* gptr) {
  asm volatile("global_load_async_to_lds_b128 %0, %1, off"
               :: "v"(lds_addr), "v"((unsigned long long)(uintptr_t)gptr)
               : "memory");
}

__device__ __forceinline__ unsigned int lds_lo32(const void* p) {
  return (unsigned int)(uintptr_t)p;   // flat LDS aperture: addr[31:0] = offset
}

// ---------------------------------------------------------------------------
// Kernel 1: fused QKV projection (unchanged layout strategy).
//   Q[s,o] = (x W_q^T + b_q) / sqrt(E)  -> bf16 row-major
//   K[s,o] =  x W_k^T + b_k             -> bf16 row-major
//   Vt[b,o,s] = x W_v^T + b_v           -> bf16 TRANSPOSED for P@V B-frags
// ---------------------------------------------------------------------------
__global__ void __launch_bounds__(128)
qkv_proj_kernel(const float* __restrict__ x,
                const float* __restrict__ Wq, const float* __restrict__ bq,
                const float* __restrict__ Wk, const float* __restrict__ bk,
                const float* __restrict__ Wv, const float* __restrict__ bv,
                __bf16* __restrict__ Qb, __bf16* __restrict__ Kb,
                __bf16* __restrict__ Vt) {
  const int lane = threadIdx.x & 31;
  const int w    = threadIdx.x >> 5;
  const int tile = blockIdx.x * 4 + w;       // 0 .. B*S/16 - 1
  const int row0 = tile * 16;
  const int n    = lane & 15;
  const int hh   = (lane >> 4) & 1;

  v16bf xa[4];
#pragma unroll
  for (int kc = 0; kc < 4; ++kc)
    xa[kc] = a_frag_f32(x + row0 * EMB, EMB, kc * 32, lane);

  const float* Ws[3] = {Wq, Wk, Wv};
  const float* bs[3] = {bq, bk, bv};
  const float qscale = 0.08838834764831845f;   // 1/sqrt(128)

#pragma unroll
  for (int mtx = 0; mtx < 3; ++mtx) {
#pragma unroll
    for (int ot = 0; ot < 8; ++ot) {
      const int o0 = ot * 16;
      const float bias = bs[mtx][o0 + n];
      v8f acc;
#pragma unroll
      for (int i = 0; i < 8; ++i) acc[i] = bias;
#pragma unroll
      for (int kc = 0; kc < 4; ++kc) {
        v16bf bw = b_frag_f32(Ws[mtx], EMB, o0, kc * 32, lane);
        acc = wmma_bf16(xa[kc], bw, acc);
      }
      if (mtx == 0) {
#pragma unroll
        for (int i = 0; i < 8; ++i) acc[i] *= qscale;
      }
#pragma unroll
      for (int i = 0; i < 8; ++i) {
        const int row = row0 + i + 8 * hh;
        const int col = o0 + n;
        const __bf16 v = (__bf16)acc[i];
        if (mtx == 0)      Qb[row * EMB + col] = v;
        else if (mtx == 1) Kb[row * EMB + col] = v;
        else {
          const int b = row >> 13;
          const int s = row & (SEQ - 1);
          Vt[(b * EMB + col) * SEQ + s] = v;
        }
      }
    }
  }
}

// ---------------------------------------------------------------------------
// Kernel 2: flash attention with async Global->LDS double-buffered K/V
// staging shared by the 4 waves of a block (same batch, same key stream).
// Per 32-key tile: 8 WMMA (QK^T) + online softmax + 8 WMMA (PV), with the
// next tile's async copies in flight during compute.
// ---------------------------------------------------------------------------
__global__ void __launch_bounds__(128)
flash_attn_kernel(const __bf16* __restrict__ Qb,
                  const __bf16* __restrict__ Kb,
                  const __bf16* __restrict__ Vt,
                  float* __restrict__ out) {
  __shared__ alignas(16) __bf16 Ks[2][32 * EMB];   // [key][e]   8 KB / buf
  __shared__ alignas(16) __bf16 Vs[2][EMB * 32];   // [e][key]   8 KB / buf
  __shared__ alignas(32) __bf16 Plds[4][16 * 32];  // per-wave P tile

  const int tid  = threadIdx.x;
  const int lane = tid & 31;
  const int w    = tid >> 5;
  const int tile = blockIdx.x * 4 + w;
  const int row0 = tile * 16;
  const int b    = row0 >> 13;                 // batch (uniform per block)
  const int n    = lane & 15;
  const int hh   = (lane >> 4) & 1;

  const __bf16* Kbase = Kb + b * SEQ * EMB;
  const __bf16* Vbase = Vt + b * EMB * SEQ;

  // Issue async copies for one 32-key tile into buffer `buf`.
  // K tile: 8 KB contiguous; V tile: 128 rows x 64 B (row stride 16 KB).
  auto issue_tile = [&](int kt, int buf) {
    const unsigned int kl = lds_lo32(&Ks[buf][0]);
    const char* kg = (const char*)(Kbase + (size_t)kt * 32 * EMB);
#pragma unroll
    for (int j = 0; j < 4; ++j)
      async_copy_b128(kl + tid * 16 + j * 2048, kg + tid * 16 + j * 2048);
    const unsigned int vl = lds_lo32(&Vs[buf][0]);
    const char* vg = (const char*)(Vbase + (size_t)tid * SEQ + kt * 32);
#pragma unroll
    for (int j = 0; j < 4; ++j)
      async_copy_b128(vl + tid * 64 + j * 16, vg + j * 16);
  };

  // resident Q fragments (softmax scale folded in at projection time)
  v16bf qa[4];
#pragma unroll
  for (int kc = 0; kc < 4; ++kc)
    qa[kc] = a_frag_bf16(Qb + row0 * EMB, EMB, kc * 32, lane);

  float mrow[8], lrow[8];
  v8f   o[8];
#pragma unroll
  for (int i = 0; i < 8; ++i) { mrow[i] = -1e30f; lrow[i] = 0.f; }
#pragma unroll
  for (int ot = 0; ot < 8; ++ot) o[ot] = (v8f){};

  __bf16* myP = &Plds[w][0];
  const int NT = SEQ / 32;                 // 256 key tiles

  issue_tile(0, 0);                        // prologue: tile 0 -> buf 0

  for (int kt = 0; kt < NT; ++kt) {
    const int cur = kt & 1;

    asm volatile("s_wait_asynccnt 0x0" ::: "memory");  // my copies of kt done
    __syncthreads();                                   // everyone's copies done
    if (kt + 1 < NT) issue_tile(kt + 1, cur ^ 1);      // overlap next copy

    const __bf16* ks = &Ks[cur][0];
    const __bf16* vs = &Vs[cur][0];

    // ---- S = Q K^T for 32 keys (operands from LDS) ----
    v8f c0 = {}, c1 = {};
#pragma unroll
    for (int kc = 0; kc < 4; ++kc) {
      v16bf bk0 = b_frag_bf16(ks, EMB, 0,  kc * 32, lane);
      c0 = wmma_bf16(qa[kc], bk0, c0);
      v16bf bk1 = b_frag_bf16(ks, EMB, 16, kc * 32, lane);
      c1 = wmma_bf16(qa[kc], bk1, c1);
    }

    // ---- online softmax (row values live in one 16-lane half) ----
    float alpha[8];
#pragma unroll
    for (int i = 0; i < 8; ++i) {
      float t = fmaxf(c0[i], c1[i]);
      t = fmaxf(t, __shfl_xor(t, 1));
      t = fmaxf(t, __shfl_xor(t, 2));
      t = fmaxf(t, __shfl_xor(t, 4));
      t = fmaxf(t, __shfl_xor(t, 8));
      const float mn = fmaxf(mrow[i], t);
      alpha[i] = __expf(mrow[i] - mn);
      mrow[i]  = mn;
      const float p0 = __expf(c0[i] - mn);
      const float p1 = __expf(c1[i] - mn);
      float rs = p0 + p1;
      rs += __shfl_xor(rs, 1);
      rs += __shfl_xor(rs, 2);
      rs += __shfl_xor(rs, 4);
      rs += __shfl_xor(rs, 8);
      lrow[i] = lrow[i] * alpha[i] + rs;
      c0[i] = p0;
      c1[i] = p1;
    }

#pragma unroll
    for (int ot = 0; ot < 8; ++ot)
#pragma unroll
      for (int i = 0; i < 8; ++i) o[ot][i] *= alpha[i];

    // ---- P: C-layout -> A-layout via per-wave LDS tile ----
#pragma unroll
    for (int i = 0; i < 8; ++i) {
      const int row = i + 8 * hh;
      myP[row * 32 + n]      = (__bf16)c0[i];
      myP[row * 32 + 16 + n] = (__bf16)c1[i];
    }
    asm volatile("s_wait_dscnt 0x0" ::: "memory");     // wave-local RAW fence

    v16bf pa;
    {
      const __bf16* p = myP + (lane & 15) * 32 + hh * 8;
      v8bf lo = *(const v8bf*)p;
      v8bf hi = *(const v8bf*)(p + 16);
#pragma unroll
      for (int j = 0; j < 8; ++j) { pa[j] = lo[j]; pa[8 + j] = hi[j]; }
    }

    // ---- O += P @ V (V B-frags from LDS: lane n -> row e=ot*16+n) ----
#pragma unroll
    for (int ot = 0; ot < 8; ++ot) {
      const __bf16* vp = vs + (ot * 16 + n) * 32 + hh * 8;
      v8bf vlo = *(const v8bf*)vp;
      v8bf vhi = *(const v8bf*)(vp + 16);
      v16bf bv;
#pragma unroll
      for (int j = 0; j < 8; ++j) { bv[j] = vlo[j]; bv[8 + j] = vhi[j]; }
      o[ot] = wmma_bf16(pa, bv, o[ot]);
    }
  }

  // ---- epilogue: divide by softmax sum, store fp32 ----
  float invl[8];
#pragma unroll
  for (int i = 0; i < 8; ++i) invl[i] = 1.0f / lrow[i];
#pragma unroll
  for (int ot = 0; ot < 8; ++ot)
#pragma unroll
    for (int i = 0; i < 8; ++i) {
      const int row = row0 + i + 8 * hh;
      const int col = ot * 16 + n;
      out[row * EMB + col] = o[ot][i] * invl[i];
    }
}

// ---------------------------------------------------------------------------
extern "C" void kernel_launch(void* const* d_in, const int* in_sizes, int n_in,
                              void* d_out, int out_size, void* d_ws, size_t ws_size,
                              hipStream_t stream) {
  const float* x  = (const float*)d_in[0];
  const float* Wq = (const float*)d_in[1];
  const float* bq = (const float*)d_in[2];
  const float* Wk = (const float*)d_in[3];
  const float* bk = (const float*)d_in[4];
  const float* Wv = (const float*)d_in[5];
  const float* bv = (const float*)d_in[6];
  float* out = (float*)d_out;

  char* ws = (char*)d_ws;
  __bf16* Qb = (__bf16*)(ws);
  __bf16* Kb = (__bf16*)(ws + (size_t)QKV_BYTES);
  __bf16* Vt = (__bf16*)(ws + (size_t)2 * QKV_BYTES);

  const int tiles  = (BATCH * SEQ) / 16;   // 1024 waves total
  const int blocks = tiles / 4;            // 4 waves (128 threads) per block

  qkv_proj_kernel<<<blocks, 128, 0, stream>>>(x, Wq, bq, Wk, bk, Wv, bv,
                                              Qb, Kb, Vt);
  flash_attn_kernel<<<blocks, 128, 0, stream>>>(Qb, Kb, Vt, out);
}
